// MultiHeadAttention_29575144800983
// MI455X (gfx1250) — matrix operand order repaired and checked by reading, not compiled, not run on hardware
//
#include <hip/hip_runtime.h>
#include <hip/hip_bf16.h>
#include <cstdint>

// ---------------------------------------------------------------------------
// Multi-head attention forward (B=2, L=2048, D=1024, H=16, d_k=64), f32 I/O.
// Compute: f16 WMMA (v_wmma_f32_16x16x32_f16) with f32 accumulation.
// Data movement: double-buffered Tensor Data Mover (tensor_load_to_lds)
// pipelined against compute via s_wait_tensorcnt.
// Layouts are chosen so EVERY fragment load is a contiguous 32B ds_load_b128
// pair: weights pre-transposed [N][K], V head-transposed [B,H,d_k,L].
// ---------------------------------------------------------------------------

typedef __attribute__((ext_vector_type(16))) _Float16     v16h;
typedef __attribute__((ext_vector_type(8)))  _Float16     v8h;
typedef __attribute__((ext_vector_type(8)))  float        v8f;
typedef __attribute__((ext_vector_type(4)))  unsigned int v4u;
typedef __attribute__((ext_vector_type(8)))  int          v8i;
typedef __attribute__((ext_vector_type(4)))  int          v4i;

#define LSEQ   2048
#define DMODEL 1024
#define NHEAD  16
#define DKH    64
#define NBATCH 2
#define MROWS  (NBATCH * LSEQ)   // 4096

#if defined(__HIP_DEVICE_COMPILE__) && __has_builtin(__builtin_amdgcn_tensor_load_to_lds)
#define HAVE_TDM 1
#else
#define HAVE_TDM 0
#endif

// Escape the LDS object into an asm with a memory clobber so the compiler
// must assume the TDM engine wrote it (otherwise the tile loads get DCE'd).
__device__ __forceinline__ void lds_acquire(const void* p) {
  asm volatile("" :: "v"(p) : "memory");
}

// ---------------- TDM: 2-D tile load (16-bit elements) ---------------------
__device__ __forceinline__ void tdm_load_2d(unsigned lds_off, const void* gptr,
                                            int td0, int td1, int stride0) {
#if HAVE_TDM
  unsigned long long ga = (unsigned long long)(uintptr_t)gptr;
  v4u g0;
  g0[0] = 1u;                                   // count=1, user descriptor
  g0[1] = lds_off;                              // lds_addr (bytes)
  g0[2] = (unsigned)(ga & 0xFFFFFFFFull);       // global_addr[31:0]
  g0[3] = (unsigned)((ga >> 32) & 0x1FFFFFFull) // global_addr[56:32]
        | 0x80000000u;                          // type=2 ("image")
  v8i g1;
  g1[0] = 1 << 16;                              // data_size = 2 bytes
  g1[1] = (td0 & 0xFFFF) << 16;                 // tensor_dim0[15:0]
  g1[2] = ((td0 >> 16) & 0xFFFF) | ((td1 & 0xFFFF) << 16);   // dim0 hi | tensor_dim1 lo
  g1[3] = ((td1 >> 16) & 0xFFFF) | ((td0 & 0xFFFF) << 16);   // dim1 hi | tile_dim0
  g1[4] = (td1 & 0xFFFF);                       // tile_dim1 (tile_dim2=0)
  g1[5] = stride0;                              // tensor_dim0_stride[31:0]
  g1[6] = 0;
  g1[7] = 0;
  v4i z4 = {0, 0, 0, 0};
#if __clang_major__ >= 23
  v8i z8 = {0, 0, 0, 0, 0, 0, 0, 0};
  __builtin_amdgcn_tensor_load_to_lds(g0, g1, z4, z4, z8, 0);
#else
  __builtin_amdgcn_tensor_load_to_lds(g0, g1, z4, z4, 0);
#endif
#else
  (void)lds_off; (void)gptr; (void)td0; (void)td1; (void)stride0;
#endif
}

template <int N>
__device__ __forceinline__ void wait_tensor() {
#if HAVE_TDM
#if __has_builtin(__builtin_amdgcn_s_wait_tensorcnt)
  __builtin_amdgcn_s_wait_tensorcnt(N);
#else
  asm volatile("s_wait_tensorcnt %0" :: "i"(N) : "memory");
#endif
#endif
}

// ---------------- WMMA fragment loaders (wave32 layouts) -------------------
// A 16x32 f16: lanes 0-15 -> M=lane, K = {0..15};
//              lanes 16-31 -> M=lane-16, K = {16..31}.
// Two contiguous 16B loads per lane.
__device__ __forceinline__ v16h load_afrag(const _Float16* s, int ld) {
  int lane = threadIdx.x & 31;
  const _Float16* row = s + (size_t)(lane & 15) * ld + ((lane >> 4) << 4);
  v8h lo = *(const v8h*)row;
  v8h hi = *(const v8h*)(row + 8);
  return __builtin_shufflevector(lo, hi, 0,1,2,3,4,5,6,7,8,9,10,11,12,13,14,15);
}

// B 32x16 f16 where column n is CONTIGUOUS in memory (s[n*ld + k]):
// lane half = column n, K = (lane>>4)*16 + e. One 32B contiguous load.
__device__ __forceinline__ v16h load_bfrag_nmajor(const _Float16* s, int ld) {
  int lane = threadIdx.x & 31;
  const _Float16* p = s + (size_t)(lane & 15) * ld + ((lane >> 4) << 4);
  return *(const v16h*)p;
}

__device__ __forceinline__ v8f wmma_f16(v16h a, v16h b, v8f c) {
  return __builtin_amdgcn_wmma_f32_16x16x32_f16(false, a, false, b, (short)0, c,
                                                false, false);
}

// ---------------- f32 -> f16 conversion ------------------------------------
__global__ __launch_bounds__(256) void cvt_f32_f16(const float* __restrict__ x,
                                                   _Float16* __restrict__ y, int n) {
  int i = blockIdx.x * blockDim.x + threadIdx.x;
  int stride = gridDim.x * blockDim.x;
  for (; i < n; i += stride) y[i] = (_Float16)x[i];
}

// f32 [K][N] -> f16 transposed [N][K] (weights; done once, LDS-tiled).
__global__ __launch_bounds__(256) void cvt_transpose_f32_f16(
    const float* __restrict__ x, _Float16* __restrict__ y) {
  __shared__ _Float16 t[32][33];
  const int bx = blockIdx.x * 32;   // n block
  const int by = blockIdx.y * 32;   // k block
  const int tx = threadIdx.x & 31;
  const int ty = threadIdx.x >> 5;  // 0..7
  for (int r = ty; r < 32; r += 8)
    t[r][tx] = (_Float16)x[(size_t)(by + r) * DMODEL + bx + tx];
  __syncthreads();
  for (int r = ty; r < 32; r += 8)
    y[(size_t)(bx + r) * DMODEL + by + tx] = t[tx][r];
}

// ---------------- WMMA GEMM: C[4096x1024] = A[4096x1024] x W + bias --------
// W passed TRANSPOSED as Wt[N][K]. Block tile 64x128, 8 waves of 32x32,
// K-stage 64, double-buffered TDM (chunk i+1 streams while chunk i computes).
// mode 0: f16 -> [B,H,L,DKH] (Q/K; Q folds the 1/8 softmax scale)
// mode 2: f16 -> [B,H,DKH,L] (V, head-transposed for the PV WMMA layout)
// mode 1: f32 row-major -> d_out (output projection)
#define GBK 64
struct GemmBuf { _Float16 As[64 * GBK]; _Float16 Bs[128 * GBK]; };  // 8KB + 16KB

__global__ __launch_bounds__(256) void wmma_gemm_kernel(
    const _Float16* __restrict__ A, const _Float16* __restrict__ Wt,
    const float* __restrict__ bias, void* __restrict__ out,
    int mode, float scale) {
  __shared__ GemmBuf sb[2];                       // 48 KB
  const int tid  = threadIdx.x;
  const int lane = tid & 31;
  const int wave = tid >> 5;
  const int wm = wave & 1, wn = wave >> 1;
  const int m0 = blockIdx.y * 64;
  const int n0 = blockIdx.x * 128;
  const unsigned BUFB = (unsigned)sizeof(GemmBuf);      // 24576
  const unsigned ASZ  = (unsigned)sizeof(sb[0].As);     // 8192

  v8f c00 = {}, c01 = {}, c10 = {}, c11 = {};

  const int NI = DMODEL / GBK;                    // 16 stages
#if HAVE_TDM
  if (tid == 0) {
    tdm_load_2d(0u,  A  + (size_t)m0 * DMODEL, GBK, 64, DMODEL);
    tdm_load_2d(ASZ, Wt + (size_t)n0 * DMODEL, GBK, 128, DMODEL);
  }
#endif
  for (int i = 0; i < NI; ++i) {
    const int cur = i & 1;
    const int kc  = i * GBK;
#if HAVE_TDM
    if (tid == 0 && i + 1 < NI) {
      const int kn = kc + GBK;
      tdm_load_2d((1 - cur) * BUFB,       A  + (size_t)m0 * DMODEL + kn, GBK, 64, DMODEL);
      tdm_load_2d((1 - cur) * BUFB + ASZ, Wt + (size_t)n0 * DMODEL + kn, GBK, 128, DMODEL);
    }
    if (tid < 32) {
      if (i + 1 < NI) wait_tensor<2>(); else wait_tensor<0>();
    }
#else
    {
      int r = tid >> 3, c = (tid & 7) * 8;        // A: 32 rows x 64 cols (rows 32..63 not loaded)
      *(v8h*)&sb[cur].As[r * GBK + c] =
          *(const v8h*)&A[(size_t)(m0 + r) * DMODEL + kc + c];
      int rb = tid >> 1, cb = (tid & 1) * 32;     // Bs: 128 n-rows x 64 k-cols
#pragma unroll
      for (int j = 0; j < 4; ++j)
        *(v8h*)&sb[cur].Bs[rb * GBK + cb + j * 8] =
            *(const v8h*)&Wt[(size_t)(n0 + rb) * DMODEL + kc + cb + j * 8];
    }
#endif
    __syncthreads();
    lds_acquire(sb);                              // force real ds_load of tiles
    const _Float16* As = sb[cur].As;
    const _Float16* Bs = sb[cur].Bs;
#pragma unroll
    for (int ks = 0; ks < GBK; ks += 32) {
      v16h a0 = load_afrag(&As[(wm * 32) * GBK + ks], GBK);
      v16h a1 = load_afrag(&As[(wm * 32 + 16) * GBK + ks], GBK);
      v16h b0 = load_bfrag_nmajor(&Bs[(wn * 32) * GBK + ks], GBK);
      v16h b1 = load_bfrag_nmajor(&Bs[(wn * 32 + 16) * GBK + ks], GBK);
      c00 = wmma_f16(a0, b0, c00);
      c01 = wmma_f16(a0, b1, c01);
      c10 = wmma_f16(a1, b0, c10);
      c11 = wmma_f16(a1, b1, c11);
    }
    __syncthreads();                              // tile fully consumed
  }

  // Epilogue. C layout: VGPR r -> rows r / r+8 (lane halves), lane&15 -> col.
  const int colL = lane & 15;
  const int rowH = (lane >> 4) << 3;
  v8f* accs[4] = { &c00, &c01, &c10, &c11 };
#pragma unroll
  for (int f = 0; f < 4; ++f) {
    int fm = f >> 1, fn = f & 1;
    int cg = n0 + wn * 32 + fn * 16 + colL;
    float bv = bias[cg];
    v8f acc = *accs[f];
#pragma unroll
    for (int r = 0; r < 8; ++r) {
      int rg = m0 + wm * 32 + fm * 16 + rowH + r;
      float val = (acc[r] + bv) * scale;
      int hh = cg >> 6, d = cg & 63;
      int bi = rg >> 11, l = rg & (LSEQ - 1);
      if (mode == 0) {        // [B,H,L,DKH]
        ((_Float16*)out)[(((size_t)(bi * NHEAD + hh) * LSEQ + l) << 6) + d] = (_Float16)val;
      } else if (mode == 2) { // [B,H,DKH,L]  (V transposed; contiguous in l)
        ((_Float16*)out)[((size_t)(bi * NHEAD + hh) * DKH + d) * LSEQ + l] = (_Float16)val;
      } else {                // f32 row-major
        ((float*)out)[(size_t)rg * DMODEL + cg] = val;
      }
    }
  }
}

// ---------------- Flash attention: per (b,h), 64 q-rows per block ----------
// K chunk in LDS: [32 keys][64 d]  (row-major from [B,H,L,DKH])
// V chunk in LDS: [64 d][32 keys]  (row-major from [B,H,DKH,L] -> PV B-frags
// are contiguous 32B loads, no scalar gathers)
struct AttnSmem {
  _Float16 Ks[2][32 * DKH];   // offsets 0, 4096
  _Float16 Vs[2][DKH * 32];   // offsets 8192, 12288
  _Float16 Ps[4][16 * 32];    // per-wave P tile (4 KB)
  float    maskadd[LSEQ];     // 0 / -1e30 per key (8 KB)
};

__global__ __launch_bounds__(128) void attn_kernel(
    const _Float16* __restrict__ Qh, const _Float16* __restrict__ Kh,
    const _Float16* __restrict__ Vt, const int* __restrict__ mask,
    _Float16* __restrict__ Oc) {
  __shared__ AttnSmem sm;
  const int tid  = threadIdx.x;
  const int lane = tid & 31;
  const int wave = tid >> 5;
  const int colL = lane & 15;
  const int rowH = (lane >> 4) << 3;
  const int bh = blockIdx.y;          // 0..31
  const int b  = bh >> 4;
  const int h  = bh & (NHEAD - 1);
  const int q0 = blockIdx.x * 64 + wave * 16;
  const unsigned KVB = 32u * DKH * (unsigned)sizeof(_Float16);  // 4096

  for (int i = tid; i < LSEQ; i += 128)
    sm.maskadd[i] = mask[b * LSEQ + i] ? 0.0f : -1e30f;

  // Q already scaled by 1/sqrt(d_k) at projection time.
  const _Float16* Qbase = Qh + ((size_t)bh * LSEQ + q0) * DKH;
  v16h qa0 = load_afrag(Qbase, DKH);       // d = 0..31
  v16h qa1 = load_afrag(Qbase + 32, DKH);  // d = 32..63

  v8f o0 = {}, o1 = {}, o2 = {}, o3 = {};
  float mrun[8], lrun[8];
#pragma unroll
  for (int r = 0; r < 8; ++r) { mrun[r] = -1e30f; lrun[r] = 0.0f; }

  const _Float16* Kbh = Kh + (size_t)bh * LSEQ * DKH;   // [key][d]
  const _Float16* Vbh = Vt + (size_t)bh * DKH * LSEQ;   // [d][key]

  const int NC = LSEQ / 32;                // 64 key chunks
#if HAVE_TDM
  if (tid == 0) {
    tdm_load_2d(0u,       Kbh, DKH, 32, DKH);    // 32 key-rows of 64 d
    tdm_load_2d(2u * KVB, Vbh, 32, DKH, LSEQ);   // 64 d-rows of 32 keys
  }
#endif
  __syncthreads();                         // maskadd ready

  for (int i = 0; i < NC; ++i) {
    const int cur = i & 1;
    const int kb  = i * 32;
#if HAVE_TDM
    if (tid == 0 && i + 1 < NC) {          // stream next chunk into other buffer
      tdm_load_2d((1 - cur) * KVB,            Kbh + (size_t)(kb + 32) * DKH, DKH, 32, DKH);
      tdm_load_2d(2u * KVB + (1 - cur) * KVB, Vbh + (kb + 32),               32, DKH, LSEQ);
    }
    if (tid < 32) {
      if (i + 1 < NC) wait_tensor<2>(); else wait_tensor<0>();
    }
#else
    {
      const v8h* gk = (const v8h*)(Kbh + (size_t)kb * DKH);
      v8h* lk = (v8h*)sm.Ks[cur];
      for (int t = tid; t < 32 * DKH / 8; t += 128) lk[t] = gk[t];
      for (int t = tid; t < DKH * 4; t += 128) {   // V: 64 rows x 4 x v8h
        int row = t >> 2, c = (t & 3) * 8;
        *(v8h*)&sm.Vs[cur][row * 32 + c] =
            *(const v8h*)&Vbh[(size_t)row * LSEQ + kb + c];
      }
    }
#endif
    __syncthreads();
    lds_acquire(&sm);                      // force real ds_load of K/V tiles
    const _Float16* Ksb = sm.Ks[cur];
    const _Float16* Vsb = sm.Vs[cur];

    if (kb + 64 < LSEQ) {                  // warm chunk i+2 into caches
      __builtin_prefetch((const char*)(Kbh + (size_t)(kb + 64) * DKH) + tid * 32, 0, 0);
      if (tid < DKH)
        __builtin_prefetch((const char*)(Vbh + (size_t)tid * LSEQ + kb + 64), 0, 0);
    }

    // S = Q . K^T : keys kb..kb+31 (two 16-col tiles), d reduced 2x32.
    v8f s0 = {}, s1 = {};
    s0 = wmma_f16(qa0, load_bfrag_nmajor(Ksb, DKH), s0);
    s0 = wmma_f16(qa1, load_bfrag_nmajor(Ksb + 32, DKH), s0);
    s1 = wmma_f16(qa0, load_bfrag_nmajor(Ksb + 16 * DKH, DKH), s1);
    s1 = wmma_f16(qa1, load_bfrag_nmajor(Ksb + 16 * DKH + 32, DKH), s1);

    const float madd0 = sm.maskadd[kb + colL];
    const float madd1 = sm.maskadd[kb + 16 + colL];

    // Online softmax: row reductions across the 16-lane halves of wave32.
#pragma unroll
    for (int r = 0; r < 8; ++r) {
      float a = s0[r] + madd0;
      float bq = s1[r] + madd1;
      float mx = fmaxf(a, bq);
      mx = fmaxf(mx, __shfl_xor(mx, 1));
      mx = fmaxf(mx, __shfl_xor(mx, 2));
      mx = fmaxf(mx, __shfl_xor(mx, 4));
      mx = fmaxf(mx, __shfl_xor(mx, 8));
      float mnew = fmaxf(mrun[r], mx);
      float p0 = __expf(a - mnew);
      float p1 = __expf(bq - mnew);
      s0[r] = p0; s1[r] = p1;
      float rs = p0 + p1;
      rs += __shfl_xor(rs, 1);
      rs += __shfl_xor(rs, 2);
      rs += __shfl_xor(rs, 4);
      rs += __shfl_xor(rs, 8);
      float alpha = __expf(mrun[r] - mnew);
      lrun[r] = lrun[r] * alpha + rs;
      mrun[r] = mnew;
      o0[r] *= alpha; o1[r] *= alpha; o2[r] *= alpha; o3[r] *= alpha;
    }

    // Reshape P: C-layout f32 -> A-layout f16 via per-wave LDS tile.
    _Float16* Pp = sm.Ps[wave];
#pragma unroll
    for (int r = 0; r < 8; ++r) {
      int row = r + rowH;
      Pp[row * 32 + colL]      = (_Float16)s0[r];
      Pp[row * 32 + 16 + colL] = (_Float16)s1[r];
    }
    asm volatile("s_wait_dscnt 0" ::: "memory");  // wave-local LDS RAW fence
    v16h pa = load_afrag(Pp, 32);

    // O += P . V : B[k=key][n=d] with column d contiguous over keys in Vs.
    o0 = wmma_f16(pa, load_bfrag_nmajor(&Vsb[0 * 16 * 32], 32), o0);
    o1 = wmma_f16(pa, load_bfrag_nmajor(&Vsb[1 * 16 * 32], 32), o1);
    o2 = wmma_f16(pa, load_bfrag_nmajor(&Vsb[2 * 16 * 32], 32), o2);
    o3 = wmma_f16(pa, load_bfrag_nmajor(&Vsb[3 * 16 * 32], 32), o3);

    __syncthreads();                       // chunk consumed; buffer reusable
  }

  // Normalize and write O in [B, L, H*DKH] so the O-projection GEMM reads it flat.
  float linv[8];
#pragma unroll
  for (int r = 0; r < 8; ++r) linv[r] = 1.0f / lrun[r];
  _Float16* Ob = Oc + ((size_t)(b * LSEQ + q0)) * DMODEL + h * DKH;
#pragma unroll
  for (int r = 0; r < 8; ++r) {
    _Float16* orow = Ob + (size_t)(r + rowH) * DMODEL;
    orow[colL]      = (_Float16)(o0[r] * linv[r]);
    orow[16 + colL] = (_Float16)(o1[r] * linv[r]);
    orow[32 + colL] = (_Float16)(o2[r] * linv[r]);
    orow[48 + colL] = (_Float16)(o3[r] * linv[r]);
  }
}

// ---------------------------------------------------------------------------
extern "C" void kernel_launch(void* const* d_in, const int* in_sizes, int n_in,
                              void* d_out, int out_size, void* d_ws, size_t ws_size,
                              hipStream_t stream) {
  (void)in_sizes; (void)n_in; (void)out_size; (void)ws_size;
  const float* q    = (const float*)d_in[0];
  const float* k    = (const float*)d_in[1];
  const float* v    = (const float*)d_in[2];
  const int*   mask = (const int*)  d_in[3];
  const float* w_q  = (const float*)d_in[4];
  const float* b_q  = (const float*)d_in[5];
  const float* w_k  = (const float*)d_in[6];
  const float* b_k  = (const float*)d_in[7];
  const float* w_v  = (const float*)d_in[8];
  const float* b_v  = (const float*)d_in[9];
  const float* w_o  = (const float*)d_in[10];
  const float* b_o  = (const float*)d_in[11];

  char* ws = (char*)d_ws;
  const size_t MB = 1ull << 20;
  _Float16* Xq = (_Float16*)(ws + 0 * MB);   // 8 MB each activation copy
  _Float16* Xk = (_Float16*)(ws + 8 * MB);
  _Float16* Xv = (_Float16*)(ws + 16 * MB);
  _Float16* Wq = (_Float16*)(ws + 24 * MB);  // 2 MB each, TRANSPOSED [N][K]
  _Float16* Wk = (_Float16*)(ws + 26 * MB);
  _Float16* Wv = (_Float16*)(ws + 28 * MB);
  _Float16* Wo = (_Float16*)(ws + 30 * MB);
  _Float16* Qh = (_Float16*)(ws + 32 * MB);  // [B,H,L,DKH] f16
  _Float16* Kh = (_Float16*)(ws + 40 * MB);  // [B,H,L,DKH] f16
  _Float16* Vt = (_Float16*)(ws + 48 * MB);  // [B,H,DKH,L] f16 (transposed)
  _Float16* Oc = (_Float16*)(ws + 56 * MB);  // [B,L,H*DKH] f16

  const int ACT = MROWS * DMODEL;     // 4,194,304
  cvt_f32_f16<<<1024, 256, 0, stream>>>(q, Xq, ACT);
  cvt_f32_f16<<<1024, 256, 0, stream>>>(k, Xk, ACT);
  cvt_f32_f16<<<1024, 256, 0, stream>>>(v, Xv, ACT);
  dim3 tgrid(DMODEL / 32, DMODEL / 32);
  cvt_transpose_f32_f16<<<tgrid, 256, 0, stream>>>(w_q, Wq);
  cvt_transpose_f32_f16<<<tgrid, 256, 0, stream>>>(w_k, Wk);
  cvt_transpose_f32_f16<<<tgrid, 256, 0, stream>>>(w_v, Wv);
  cvt_transpose_f32_f16<<<tgrid, 256, 0, stream>>>(w_o, Wo);

  dim3 ggrid(DMODEL / 128, MROWS / 64);   // (8, 64)
  // Q projection folds the 1/sqrt(d_k)=0.125 softmax scale into the store.
  wmma_gemm_kernel<<<ggrid, 256, 0, stream>>>(Xq, Wq, b_q, Qh, 0, 0.125f);
  wmma_gemm_kernel<<<ggrid, 256, 0, stream>>>(Xk, Wk, b_k, Kh, 0, 1.0f);
  wmma_gemm_kernel<<<ggrid, 256, 0, stream>>>(Xv, Wv, b_v, Vt, 2, 1.0f);

  attn_kernel<<<dim3(LSEQ / 64, NBATCH * NHEAD), 128, 0, stream>>>(Qh, Kh, Vt, mask, Oc);

  wmma_gemm_kernel<<<ggrid, 256, 0, stream>>>(Oc, Wo, b_o, d_out, 1, 1.0f);
}
